// LorentzKG_86337432584671
// MI455X (gfx1250) — matrix operand order repaired and checked by reading, not compiled, not run on hardware
//
#include <hip/hip_runtime.h>
#include <math.h>

// Problem constants (from reference)
#define N_ENT  40000
#define RANK   200
#define BQ     500
#define KDIM   1000   // RANK * 5
#define KPAD   1024   // K padded to multiple of 32 (WMMA K-step)
#define MPAD   512    // M padded to multiple of 16
#define NKT    32     // K chunks of 32
#define LDSROW 40     // padded LDS row stride in halves (80 B = 20 banks)
#define SCORES_ELEMS ((size_t)BQ * N_ENT)

typedef __attribute__((ext_vector_type(16))) _Float16 v16h;
typedef __attribute__((ext_vector_type(8)))  _Float16 v8h;
typedef __attribute__((ext_vector_type(8)))  float    v8f;

// ---------------------------------------------------------------------------
// Kernel 0: zero the f16 A-staging buffer (padding rows/cols must be 0 so the
// K/M tail WMMAs contribute nothing).
// ---------------------------------------------------------------------------
__global__ void lkg_zero_kernel(uint4* __restrict__ ws) {
    int i = blockIdx.x * blockDim.x + threadIdx.x;   // 65536 uint4 = 1 MB
    uint4 z; z.x = 0u; z.y = 0u; z.z = 0u; z.w = 0u;
    ws[i] = z;
}

// ---------------------------------------------------------------------------
// Kernel 1: gather + lorentz_rotate (exact GELU) + lorentz_boost2 + negate t.
// One thread per (b, d) 5-vector. Writes A_f16[b][d*5 + i] and reg_rel.
// ---------------------------------------------------------------------------
__global__ void lkg_prep_kernel(const int*   __restrict__ queries,
                                const float* __restrict__ ent,
                                const float* __restrict__ rot_tab,
                                const float* __restrict__ boo_tab,
                                _Float16*    __restrict__ Abuf,
                                float*       __restrict__ reg_out) {
    int idx = blockIdx.x * blockDim.x + threadIdx.x;
    if (idx >= BQ * RANK) return;
    int b = idx / RANK;
    int d = idx - b * RANK;

    int ei = queries[b * 3 + 0];
    int ri = queries[b * 3 + 1];

    const float* lp = ent     + ((size_t)ei * RANK + d) * 5;
    const float* rp = rot_tab + ((size_t)ri * RANK + d) * 4;
    const float* bp = boo_tab + ((size_t)ri * RANK + d) * 4;

    float t  = lp[0], qr = lp[1], qi = lp[2], qj = lp[3], qk = lp[4];

    float rotv[4], boov[4];
    float reg_r = 0.0f, reg_b = 0.0f;
#pragma unroll
    for (int m = 0; m < 4; ++m) {
        rotv[m] = rp[m];  boov[m] = bp[m];
        reg_r += rotv[m] * rotv[m];
        reg_b += boov[m] * boov[m];
    }
    reg_out[(size_t)b * RANK + d] = sqrtf(reg_r) + sqrtf(reg_b);

    // exact GELU: 0.5 * x * (1 + erf(x / sqrt(2)))
    float g[4];
#pragma unroll
    for (int m = 0; m < 4; ++m)
        g[m] = 0.5f * rotv[m] * (1.0f + erff(rotv[m] * 0.7071067811865475f));
    float rr = g[0], rI = g[1], rJ = g[2], rK = g[3];

    // quaternion-style rotate
    float sA = qr * rr - qi * rI - qj * rJ - qk * rK;
    float sB = qr * rI + rr * qi + qj * rK - rJ * qk;
    float sC = qr * rJ + rr * qj + qk * rI - rK * qi;
    float sD = qr * rK + rr * qk + qi * rJ - rI * qj;
    float s[4] = { sA, sB, sC, sD };

    // boost: b = sigmoid(boost) / 4
    float bs[4], b2 = 0.0f;
#pragma unroll
    for (int m = 0; m < 4; ++m) {
        bs[m] = 0.25f / (1.0f + expf(-boov[m]));
        b2 += bs[m] * bs[m];
    }
    float xt = t * b2;
#pragma unroll
    for (int m = 0; m < 4; ++m) xt += bs[m] * s[m];

    float xs[4];
#pragma unroll
    for (int i2 = 0; i2 < 4; ++i2) {
        float acc = t * bs[i2];
#pragma unroll
        for (int j2 = 0; j2 < 4; ++j2) {
            float eye = (i2 == j2) ? 1.0f : 0.0f;
            acc += sqrtf(eye + bs[i2] * bs[j2]) * s[j2];
        }
        xs[i2] = acc;
    }

    // x = lhs with time component negated
    _Float16* ap = Abuf + (size_t)b * KPAD + d * 5;
    ap[0] = (_Float16)(-xt);
#pragma unroll
    for (int m = 0; m < 4; ++m) ap[m + 1] = (_Float16)xs[m];
}

// ---------------------------------------------------------------------------
// Helper: pack 8 floats (two float4) into v8h
// ---------------------------------------------------------------------------
__device__ __forceinline__ v8h pack_f16x8(float4 f0, float4 f1) {
    v8h h;
    h[0] = (_Float16)f0.x; h[1] = (_Float16)f0.y;
    h[2] = (_Float16)f0.z; h[3] = (_Float16)f0.w;
    h[4] = (_Float16)f1.x; h[5] = (_Float16)f1.y;
    h[6] = (_Float16)f1.z; h[7] = (_Float16)f1.w;
    return h;
}

// ---------------------------------------------------------------------------
// Kernel 2: WMMA GEMM  scores[b][n] = 2 + (A[b,:] . ent[n,:]) / 100
//
// Block: 8 waves, 128(M) x 64(N) tile; wave: 16(M) x 64(N) with 4 v8f accs.
// Shared-B double-buffered LDS staging: 256 threads cooperatively load the
// 64(n) x 32(k) f32 ent tile, convert to f16 ONCE, store to LDS (row stride
// 40 halves to spread banks). Chunk kt+1 global loads issue before the
// WMMAs of chunk kt; the wait lands after them (latency hidden).
//
// Fragment layouts (wave32):
//  A 16x32 f16: lane half h, m = lane&15, elem e -> K=(e>>3)*16 + h*8 + (e&7)
//  B 32x16 f16: lane half h, n = lane&15, elem e -> K=(e>>3)*16 + h*8 + (e&7)
//  C 16x16 f32: elem r -> M = 8*h + r, N = lane&15
// ---------------------------------------------------------------------------
__global__ void __launch_bounds__(256)
lkg_gemm_kernel(const _Float16* __restrict__ Abuf,
                const float*    __restrict__ ent,
                float*          __restrict__ out) {
    __shared__ _Float16 sB[2][64 * LDSROW];   // 2 x 5 KB

    const int tid     = threadIdx.x;
    const int lane    = tid & 31;
    const int wave    = tid >> 5;
    const int halfsel = lane >> 4;
    const int mrow    = lane & 15;
    const int ncol    = lane & 15;

    const int m0 = (blockIdx.y * 8 + wave) * 16;
    const int n0 = blockIdx.x * 64;

    // staging role: 4 threads per n-row, 8 contiguous k each
    const int n_local = tid >> 2;            // 0..63
    const int k_grp   = (tid & 3) * 8;       // 0,8,16,24
    const float*    srow = ent + (size_t)(n0 + n_local) * KDIM + k_grp;
    _Float16* const sdst = (_Float16*)&sB[0][0] + n_local * LDSROW + k_grp;

    v8f acc[4];
#pragma unroll
    for (int j = 0; j < 4; ++j)
#pragma unroll
        for (int r = 0; r < 8; ++r) acc[j][r] = 0.0f;

    const _Float16* arow = Abuf + (size_t)(m0 + mrow) * KPAD + halfsel * 8;

    // ---- prologue: stage chunk 0 (k 0..31, always in-bounds) ----
    {
        float4 f0 = *(const float4*)(srow);
        float4 f1 = *(const float4*)(srow + 4);
        *(v8h*)sdst = pack_f16x8(f0, f1);
    }
    __syncthreads();

    for (int kt = 0; kt < NKT; ++kt) {
        const int cur = kt & 1;
        const int k0  = kt * 32;
        const bool have_next = (kt + 1 < NKT);

        // ---- issue next chunk's global loads (no wait yet) ----
        float4 g0 = make_float4(0.f, 0.f, 0.f, 0.f);
        float4 g1 = g0;
        if (have_next) {
            const int kn = k0 + 32;
            if (kn + k_grp < KDIM) {          // K tail: zero-fill beyond 999
                g0 = *(const float4*)(srow + kn);
                g1 = *(const float4*)(srow + kn + 4);
            }
            if (kt + 2 < NKT)                 // deepen pipe: prefetch kt+2
                __builtin_prefetch((const void*)(srow + kn + 32), 0, 1);
        }

        // ---- A fragment (global, 1 MB hot buffer -> cache hits) ----
        v8h alo = *(const v8h*)(arow + k0);
        v8h ahi = *(const v8h*)(arow + k0 + 16);
        v16h a  = __builtin_shufflevector(alo, ahi,
                    0,1,2,3,4,5,6,7,8,9,10,11,12,13,14,15);

        // ---- 4 WMMAs against shared-B from LDS ----
#pragma unroll
        for (int j = 0; j < 4; ++j) {
            const _Float16* bp = &sB[cur][(j * 16 + ncol) * LDSROW + halfsel * 8];
            v8h b0 = *(const v8h*)(bp);
            v8h b1 = *(const v8h*)(bp + 16);
            v16h bf = __builtin_shufflevector(b0, b1,
                        0,1,2,3,4,5,6,7,8,9,10,11,12,13,14,15);
            acc[j] = __builtin_amdgcn_wmma_f32_16x16x32_f16(
                         false, a, false, bf, (short)0, acc[j], false, false);
        }

        // ---- convert + store next chunk into the other buffer ----
        if (have_next)
            *(v8h*)(sdst + (1 - cur) * 64 * LDSROW) = pack_f16x8(g0, g1);
        __syncthreads();
    }

    // ---- epilogue: scores = 2 + 2*cin, cin = dot/200  =>  2 + acc/100 ----
#pragma unroll
    for (int j = 0; j < 4; ++j) {
        const int n = n0 + j * 16 + ncol;
#pragma unroll
        for (int r = 0; r < 8; ++r) {
            int m = m0 + halfsel * 8 + r;
            if (m < BQ)
                out[(size_t)m * N_ENT + n] = 2.0f + 0.01f * acc[j][r];
        }
    }
}

// ---------------------------------------------------------------------------
extern "C" void kernel_launch(void* const* d_in, const int* in_sizes, int n_in,
                              void* d_out, int out_size, void* d_ws, size_t ws_size,
                              hipStream_t stream) {
    const int*   queries = (const int*)  d_in[0];
    const float* ent     = (const float*)d_in[1];
    const float* rot     = (const float*)d_in[2];
    const float* boo     = (const float*)d_in[3];

    float*     out    = (float*)d_out;
    float*     regout = out + SCORES_ELEMS;           // reg_rel after scores
    _Float16*  Abuf   = (_Float16*)d_ws;              // 512 x 1024 f16 = 1 MB

    // 1) zero A staging (padding must be zero)
    {
        int n16 = (MPAD * KPAD * 2) / 16;             // 65536 uint4
        lkg_zero_kernel<<<n16 / 256, 256, 0, stream>>>((uint4*)d_ws);
    }
    // 2) gather + rotate + boost -> A f16, reg_rel -> out tail
    {
        int total = BQ * RANK;
        lkg_prep_kernel<<<(total + 255) / 256, 256, 0, stream>>>(
            queries, ent, rot, boo, Abuf, regout);
    }
    // 3) WMMA GEMM -> scores
    {
        dim3 grid(N_ENT / 64, MPAD / 128);            // 625 x 4 blocks
        lkg_gemm_kernel<<<grid, 256, 0, stream>>>(Abuf, ent, out);
    }
}
